// Encoder_16269336117443
// MI455X (gfx1250) — matrix operand order repaired and checked by reading, not compiled, not run on hardware
//
#include <hip/hip_runtime.h>

// ---------------------------------------------------------------------------
// Bidirectional masked LSTM encoder for MI455X (gfx1250, wave32, WMMA bf16)
//   V=50000 E=300 H=256 T=512 B=64
// ---------------------------------------------------------------------------

typedef __bf16 bf16_t;
typedef __bf16  v16bf __attribute__((ext_vector_type(16)));
typedef __bf16  v8bf  __attribute__((ext_vector_type(8)));
typedef float   v8f   __attribute__((ext_vector_type(8)));
typedef unsigned int u32x4 __attribute__((ext_vector_type(4)));

#define T_LEN 512
#define B_SZ  64
#define H_SZ  256
#define E_SZ  300
#define EPAD  320   // E padded to multiple of 32 for K loop
#define FOURH 1024

// workspace byte offsets
#define WS_XPROJ   ((size_t)0)                               // 2 * T*4*64*256 bf16 = 134217728 B
#define WS_UT      ((size_t)134217728)                       // 2*1024*256 bf16 = 1048576 B
#define WS_WT      (WS_UT + (size_t)2*1024*256*2)            // 2*1024*320 bf16 = 1310720 B
#define WS_MASKB   (WS_WT + (size_t)2*1024*320*2)            // 512 u64

// output float offsets
#define OUT_ENC    ((size_t)0)                               // [64][512][512]
#define OUT_SH     ((size_t)B_SZ*T_LEN*512)                  // [64][512]
#define OUT_SC     (OUT_SH + (size_t)B_SZ*512)
#define OUT_MASK   (OUT_SC + (size_t)B_SZ*512)               // [64][512]

// ---------------------------------------------------------------------------
// activations: prefer native gfx1250 v_tanh_f32 (1 trans op) when available
// ---------------------------------------------------------------------------
__device__ __forceinline__ float dev_tanh(float x) {
#if __has_builtin(__builtin_amdgcn_tanhf)
    return __builtin_amdgcn_tanhf(x);
#elif __has_builtin(__builtin_amdgcn_tanh_f32)
    return __builtin_amdgcn_tanh_f32(x);
#else
    x = fminf(15.0f, fmaxf(-15.0f, x));
    float e = __expf(2.0f * x);
    return (e - 1.0f) / (e + 1.0f);
#endif
}
__device__ __forceinline__ float dev_sigm(float x) {
#if __has_builtin(__builtin_amdgcn_tanhf) || __has_builtin(__builtin_amdgcn_tanh_f32)
    return fmaf(0.5f, dev_tanh(0.5f * x), 0.5f);   // sigmoid(x) = (1+tanh(x/2))/2
#else
    return 1.0f / (1.0f + __expf(-x));
#endif
}

union Frag32 { v16bf v; u32x4 q[2]; };
union Frag16 { v8bf  v; u32x4 q;    };

// ---------------------------------------------------------------------------
// prep kernels: transpose U/W to [N][K] bf16, build mask bitmask + mask output
// ---------------------------------------------------------------------------
__global__ void prep_ut(const float* __restrict__ Ufw, const float* __restrict__ Ubw,
                        bf16_t* __restrict__ Ut) {
    int idx = blockIdx.x * blockDim.x + threadIdx.x;      // 2*1024*256
    if (idx >= 2 * 1024 * 256) return;
    int d   = idx >> 18;
    int rem = idx & 262143;
    int n   = rem >> 8;
    int k   = rem & 255;
    const float* U = d ? Ubw : Ufw;
    Ut[idx] = (bf16_t)U[(size_t)k * FOURH + n];
}

__global__ void prep_wt(const float* __restrict__ Wfw, const float* __restrict__ Wbw,
                        bf16_t* __restrict__ Wt) {
    int idx = blockIdx.x * blockDim.x + threadIdx.x;      // 2*1024*320
    if (idx >= 2 * 1024 * EPAD) return;
    int d   = idx / (1024 * EPAD);
    int rem = idx % (1024 * EPAD);
    int n   = rem / EPAD;
    int k   = rem % EPAD;
    const float* W = d ? Wbw : Wfw;
    Wt[idx] = (bf16_t)((k < E_SZ) ? W[(size_t)k * FOURH + n] : 0.0f);
}

__global__ void prep_mask(const int* __restrict__ seq,
                          unsigned long long* __restrict__ maskbits,
                          float* __restrict__ maskout) {
    int idx = blockIdx.x * blockDim.x + threadIdx.x;      // B*T
    if (idx >= B_SZ * T_LEN) return;
    int b = idx >> 9;
    int t = idx & 511;
    maskout[idx] = (seq[idx] != 0) ? 1.0f : 0.0f;         // layout [B][T]
    if (b == 0) {
        unsigned long long bits = 0ull;
        for (int bb = 0; bb < 64; ++bb)
            if (seq[bb * T_LEN + t] != 0) bits |= (1ull << bb);
        maskbits[t] = bits;
    }
}

// ---------------------------------------------------------------------------
// Phase 1: embedding gather + x@W+b, output pre-swizzled bf16 tiles
//   grid (T, 4 n-chunks, 2 dirs), block 512 (16 waves)
//   each block: one timestep t, 64 batches, 256 output cols, one direction
// ---------------------------------------------------------------------------
__global__ __launch_bounds__(512)
void embed_project(const int* __restrict__ seq, const float* __restrict__ emb,
                   const float* __restrict__ b_fw, const float* __restrict__ b_bw,
                   const bf16_t* __restrict__ Wt, bf16_t* __restrict__ xproj) {
    const int t   = blockIdx.x;
    const int ny  = blockIdx.y;
    const int dir = blockIdx.z;
    const int tid = threadIdx.x;
    const int w   = tid >> 5;
    const int l   = tid & 31;
    const int lh  = l >> 4;
    const int lm  = l & 15;

    __shared__ int    stok[64];
    __shared__ bf16_t sa[64][328];      // pitch 328: bank-conflict-free A frags

    if (tid < 64) stok[tid] = seq[tid * T_LEN + t];
    __syncthreads();

    for (int idx = tid; idx < 64 * 328; idx += 512) {
        int b  = idx / 328;
        int kk = idx - b * 328;
        float v = 0.0f;
        if (kk < E_SZ) v = emb[(size_t)stok[b] * E_SZ + kk];
        sa[b][kk] = (bf16_t)v;
    }
    __syncthreads();

    const bf16_t* Wt_d = Wt + (size_t)dir * 1024 * EPAD;
    const float*  bias = dir ? b_bw : b_fw;
    bf16_t* xp_d = xproj + (size_t)dir * ((size_t)T_LEN * 4 * 64 * 256);

    const int mb     = w & 3;
    const int ntBase = (w >> 2) * 4;

    v8f acc[4];
    #pragma unroll
    for (int i = 0; i < 4; ++i) {
        float bv = bias[ny * 256 + (ntBase + i) * 16 + lm];
        #pragma unroll
        for (int r = 0; r < 8; ++r) acc[i][r] = bv;
    }

    for (int k = 0; k < EPAD / 32; ++k) {
        Frag32 afr, bfr[4];
        // fetch all 4 B fragments first (single clause, one wait)
        #pragma unroll
        for (int i = 0; i < 4; ++i) {
            const int colRow = ny * 256 + (ntBase + i) * 16 + lm;   // N index
            const int kbb    = k * 32 + lh * 16;
            const u32x4* bp = (const u32x4*)(Wt_d + (size_t)colRow * EPAD + kbb);
            bfr[i].q[0] = bp[0];
            bfr[i].q[1] = bp[1];
        }
        const int row = mb * 16 + lm;
        const int kb  = k * 32 + lh * 16;
        afr.q[0] = *(const u32x4*)&sa[row][kb];
        afr.q[1] = *(const u32x4*)&sa[row][kb + 8];
        #pragma unroll
        for (int i = 0; i < 4; ++i)
            acc[i] = __builtin_amdgcn_wmma_f32_16x16x32_bf16(
                false, afr.v, false, bfr[i].v, (short)0, acc[i], false, false);
    }

    #pragma unroll
    for (int i = 0; i < 4; ++i) {
        const int ntG = ny * 16 + ntBase + i;
        Frag16 st;
        #pragma unroll
        for (int r = 0; r < 8; ++r) st.v[r] = (bf16_t)acc[i][r];
        *(u32x4*)(xp_d + (((size_t)(t * 4 + mb) * 64 + ntG) * 256 + l * 8)) = st.q;
    }
}

// ---------------------------------------------------------------------------
// Phase 2: recurrence. grid(2) = direction, block 1024 (32 waves, one WGP).
//   h (bf16) lives in LDS; c and h carry state live in registers.
//   wave w: jb = w&15 (hidden tile), batch tiles {w>>4, (w>>4)+2}, all 4 gates.
// ---------------------------------------------------------------------------
__global__ __launch_bounds__(1024)
void lstm_recurrent(const bf16_t* __restrict__ xproj,
                    const bf16_t* __restrict__ Ut,
                    const unsigned long long* __restrict__ maskbits,
                    float* __restrict__ out) {
    const int dir = blockIdx.x;
    const int tid = threadIdx.x;
    const int w   = tid >> 5;
    const int l   = tid & 31;
    const int lh  = l >> 4;
    const int lm  = l & 15;
    const int jb  = w & 15;
    const int mbA = w >> 4;                 // tasks: mbA and mbA+2

    const bf16_t* xp_d = xproj + (size_t)dir * ((size_t)T_LEN * 4 * 64 * 256);
    const bf16_t* Ut_d = Ut + (size_t)dir * 1024 * 256;

    __shared__ bf16_t sh[64][264];          // h matrix, pitch 264: conflict-free
    for (int i = tid; i < 64 * 264; i += 1024) ((bf16_t*)sh)[i] = (bf16_t)0.0f;
    __syncthreads();

    v8f   acc[2][4];
    float creg[2][8], hreg[2][8];
    #pragma unroll
    for (int tk = 0; tk < 2; ++tk)
        #pragma unroll
        for (int r = 0; r < 8; ++r) { creg[tk][r] = 0.0f; hreg[tk][r] = 0.0f; }

    for (int step = 0; step < T_LEN; ++step) {
        const int t = dir ? (T_LEN - 1 - step) : step;
        const unsigned long long mb64 = maskbits[t];

        // prefetch next step's x_proj tiles into L2
        if (step + 1 < T_LEN) {
            const int tn = dir ? (T_LEN - 2 - step) : (step + 1);
            __builtin_prefetch(xp_d + ((size_t)(tn * 4 + mbA) * 64 + jb) * 256 + l * 8, 0, 0);
        }

        // acc = x_proj tile (pre-swizzled, matches C/D layout exactly)
        #pragma unroll
        for (int tk = 0; tk < 2; ++tk) {
            const int mb = mbA + tk * 2;
            #pragma unroll
            for (int g = 0; g < 4; ++g) {
                const int nt = g * 16 + jb;
                Frag16 xp;
                xp.q = *(const u32x4*)(xp_d + ((size_t)(t * 4 + mb) * 64 + nt) * 256 + l * 8);
                #pragma unroll
                for (int r = 0; r < 8; ++r) acc[tk][g][r] = (float)xp.v[r];
            }
        }

        // z += h @ U   (K = 256 in 8 steps of 32)
        #pragma unroll
        for (int k = 0; k < 8; ++k) {
            Frag32 afr[2], bfr[4];
            // issue all 4 gate B-fragments up front: one 8-load clause, one wait
            #pragma unroll
            for (int g = 0; g < 4; ++g) {
                const int rowN = (g * 16 + jb) * 16 + lm;
                const int kbb  = k * 32 + lh * 16;
                const u32x4* bp = (const u32x4*)(Ut_d + (size_t)rowN * 256 + kbb);
                bfr[g].q[0] = bp[0];
                bfr[g].q[1] = bp[1];
            }
            #pragma unroll
            for (int tk = 0; tk < 2; ++tk) {
                const int row = (mbA + tk * 2) * 16 + lm;
                const int kb  = k * 32 + lh * 16;
                afr[tk].q[0] = *(const u32x4*)&sh[row][kb];
                afr[tk].q[1] = *(const u32x4*)&sh[row][kb + 8];
            }
            // 8 back-to-back WMMAs, B fragments reused across both batch tiles
            #pragma unroll
            for (int g = 0; g < 4; ++g) {
                acc[0][g] = __builtin_amdgcn_wmma_f32_16x16x32_bf16(
                    false, afr[0].v, false, bfr[g].v, (short)0, acc[0][g], false, false);
                acc[1][g] = __builtin_amdgcn_wmma_f32_16x16x32_bf16(
                    false, afr[1].v, false, bfr[g].v, (short)0, acc[1][g], false, false);
            }
        }

        __syncthreads();   // all waves finished reading h(t-1)

        // gates + state update (pure per-lane: C layout m=r+8*lh, n=lm)
        #pragma unroll
        for (int tk = 0; tk < 2; ++tk) {
            const int mb = mbA + tk * 2;
            #pragma unroll
            for (int r = 0; r < 8; ++r) {
                const int m = mb * 16 + lh * 8 + r;
                const int j = jb * 16 + lm;
                float iv = dev_sigm(acc[tk][0][r]);
                float fv = dev_sigm(acc[tk][1][r]);
                float gv = dev_tanh(acc[tk][2][r]);
                float ov = dev_sigm(acc[tk][3][r]);
                float cn = fv * creg[tk][r] + iv * gv;
                float hn = ov * dev_tanh(cn);
                const bool msk = (mb64 >> m) & 1ull;
                float h2 = msk ? hn : hreg[tk][r];
                float c2 = msk ? cn : creg[tk][r];
                hreg[tk][r] = h2;
                creg[tk][r] = c2;
                sh[m][j] = (bf16_t)h2;
                out[OUT_ENC + ((size_t)m * T_LEN + t) * 512 + dir * 256 + j] = h2;
            }
        }
        __syncthreads();   // h(t) visible to all waves
    }

    // final states
    #pragma unroll
    for (int tk = 0; tk < 2; ++tk) {
        const int mb = mbA + tk * 2;
        #pragma unroll
        for (int r = 0; r < 8; ++r) {
            const int m = mb * 16 + lh * 8 + r;
            const int j = jb * 16 + lm;
            out[OUT_SH + (size_t)m * 512 + dir * 256 + j] = hreg[tk][r];
            out[OUT_SC + (size_t)m * 512 + dir * 256 + j] = creg[tk][r];
        }
    }
}

// ---------------------------------------------------------------------------
extern "C" void kernel_launch(void* const* d_in, const int* in_sizes, int n_in,
                              void* d_out, int out_size, void* d_ws, size_t ws_size,
                              hipStream_t stream) {
    const int*   seq   = (const int*)d_in[0];
    const float* emb   = (const float*)d_in[3];
    const float* W_fw  = (const float*)d_in[4];
    const float* U_fw  = (const float*)d_in[5];
    const float* b_fw  = (const float*)d_in[6];
    const float* W_bw  = (const float*)d_in[7];
    const float* U_bw  = (const float*)d_in[8];
    const float* b_bw  = (const float*)d_in[9];
    float* out = (float*)d_out;

    char* ws = (char*)d_ws;
    bf16_t* xproj = (bf16_t*)(ws + WS_XPROJ);
    bf16_t* Ut    = (bf16_t*)(ws + WS_UT);
    bf16_t* Wt    = (bf16_t*)(ws + WS_WT);
    unsigned long long* maskb = (unsigned long long*)(ws + WS_MASKB);

    (void)in_sizes; (void)n_in; (void)out_size; (void)ws_size;

    prep_ut<<<dim3((2 * 1024 * 256 + 255) / 256), dim3(256), 0, stream>>>(U_fw, U_bw, Ut);
    prep_wt<<<dim3((2 * 1024 * EPAD + 255) / 256), dim3(256), 0, stream>>>(W_fw, W_bw, Wt);
    prep_mask<<<dim3((B_SZ * T_LEN + 255) / 256), dim3(256), 0, stream>>>(
        seq, maskb, out + OUT_MASK);

    embed_project<<<dim3(T_LEN, 4, 2), dim3(512), 0, stream>>>(
        seq, emb, b_fw, b_bw, Wt, xproj);

    lstm_recurrent<<<dim3(2), dim3(1024), 0, stream>>>(xproj, Ut, maskb, out);
}
